// PBN_3917010174807
// MI455X (gfx1250) — compile-verified, matrix operand order checked by reading, not run on hardware
//
#include <hip/hip_runtime.h>
#include <hip/hip_bf16.h>
#include <math.h>

#define B_    1024
#define IND   2048
#define D_    256
#define H_    8
#define HD_   32
#define K_    448
#define OUT_  128
#define LOOPS 3

typedef __bf16 bf16_t;
typedef bf16_t v16bf __attribute__((ext_vector_type(16)));
typedef float  v8f   __attribute__((ext_vector_type(8)));

union Frag   { v16bf v; uint4 q[2]; };
union Chunk8 { uint4 q; bf16_t e[8]; };

// Load a 16-element bf16 WMMA fragment from two contiguous 16B chunks.
// Matches the CDNA5 16-bit A/B layout: lane holds K = {c0..c0+7, c1..c1+7}.
__device__ __forceinline__ v16bf frag_ld(const bf16_t* p0, const bf16_t* p1) {
    Frag f;
    f.q[0] = *reinterpret_cast<const uint4*>(p0);
    f.q[1] = *reinterpret_cast<const uint4*>(p1);
    return f.v;
}

// Intra-wave LDS fence: lanes exchanged data through LDS within one wave.
__device__ __forceinline__ void wave_lds_fence() {
    asm volatile("s_wait_dscnt 0" ::: "memory");
}

__device__ __forceinline__ float gelu_erf(float z) {
    return 0.5f * z * (1.0f + erff(z * 0.70710678118654752f));
}

__device__ __forceinline__ float block_reduce_sum(float v, float* red) {
    const int t = threadIdx.x;
    red[t] = v; __syncthreads();
    for (int s = 128; s > 0; s >>= 1) {
        if (t < s) red[t] += red[t + s];
        __syncthreads();
    }
    float r = red[0]; __syncthreads();
    return r;
}

// ---------------------------------------------------------------------------
// Weight conversion: basis^T -> bf16, wk -> bf16, pe_w2 -> bf16
// ---------------------------------------------------------------------------
__global__ void k_convert(const float* __restrict__ basis,
                          const float* __restrict__ attn_in_w,
                          const float* __restrict__ pe_w2,
                          bf16_t* __restrict__ basisT,
                          bf16_t* __restrict__ wk_bf,
                          bf16_t* __restrict__ pew2_bf) {
    const int total = IND * D_ + D_ * D_ + D_ * (D_ / 2);
    for (int i = blockIdx.x * blockDim.x + threadIdx.x; i < total;
         i += gridDim.x * blockDim.x) {
        if (i < IND * D_) {
            int n = i / IND, k = i % IND;           // basisT[n][k] = basis[k][n]
            basisT[i] = (bf16_t)basis[k * D_ + n];
        } else if (i < IND * D_ + D_ * D_) {
            int j = i - IND * D_;                    // wk = attn_in_w rows 256..511
            wk_bf[j] = (bf16_t)attn_in_w[D_ * D_ + j];
        } else {
            int j = i - IND * D_ - D_ * D_;
            pew2_bf[j] = (bf16_t)pe_w2[j];
        }
    }
}

// ---------------------------------------------------------------------------
// xl = log1p(relu(x)); keep fp32 (for top-k) and bf16 (for WMMA A)
// ---------------------------------------------------------------------------
__global__ void k_prep(const float* __restrict__ x, float* __restrict__ xl,
                       bf16_t* __restrict__ xlb) {
    for (int i = blockIdx.x * blockDim.x + threadIdx.x; i < B_ * IND;
         i += gridDim.x * blockDim.x) {
        float v = log1pf(fmaxf(x[i], 0.0f));
        xl[i] = v;
        xlb[i] = (bf16_t)v;
    }
}

// ---------------------------------------------------------------------------
// state = tanh(xl @ basis)  [1024x2048x256 bf16 WMMA]; also S = 1
// ---------------------------------------------------------------------------
__global__ __launch_bounds__(256) void k_turbo(const bf16_t* __restrict__ xlb,
                                               const bf16_t* __restrict__ basisT,
                                               float* __restrict__ state,
                                               float* __restrict__ S) {
    const int lane = threadIdx.x & 31, w = threadIdx.x >> 5;
    const int ln = lane & 15, hi = lane >> 4;
    const int m0 = blockIdx.x * 16;
    const int n0a = w * 16, n0b = (w + 8) * 16;
    v8f acc_a, acc_b;
#pragma unroll
    for (int r = 0; r < 8; ++r) { acc_a[r] = 0.0f; acc_b[r] = 0.0f; }
    const bf16_t* arow = xlb + (m0 + ln) * IND;
    const bf16_t* browA = basisT + (n0a + ln) * IND;
    const bf16_t* browB = basisT + (n0b + ln) * IND;
    for (int k0 = 0; k0 < IND; k0 += 32) {
        const int c0 = k0 + hi * 8, c1 = k0 + 16 + hi * 8;
        v16bf a  = frag_ld(arow + c0, arow + c1);
        v16bf ba = frag_ld(browA + c0, browA + c1);
        acc_a = __builtin_amdgcn_wmma_f32_16x16x32_bf16(false, a, false, ba,
                                                        (short)0, acc_a, false, false);
        v16bf bb = frag_ld(browB + c0, browB + c1);
        acc_b = __builtin_amdgcn_wmma_f32_16x16x32_bf16(false, a, false, bb,
                                                        (short)0, acc_b, false, false);
    }
#pragma unroll
    for (int r = 0; r < 8; ++r) {
        const int row = m0 + r + hi * 8;
        state[row * D_ + n0a + ln] = tanhf(acc_a[r]);
        state[row * D_ + n0b + ln] = tanhf(acc_b[r]);
        S[row * D_ + n0a + ln] = 1.0f;
        S[row * D_ + n0b + ln] = 1.0f;
    }
}

// ---------------------------------------------------------------------------
// Per-row bitonic top-448 of 2048 (descending, ties -> lower index)
// ---------------------------------------------------------------------------
__global__ __launch_bounds__(256) void k_topk(const float* __restrict__ xl,
                                              float* __restrict__ tval,
                                              int* __restrict__ tidx) {
    __shared__ float sv[IND];
    __shared__ int   si[IND];
    const int b = blockIdx.x, t = threadIdx.x;
    for (int i = t; i < IND; i += 256) { sv[i] = xl[b * IND + i]; si[i] = i; }
    __syncthreads();
    for (int size = 2; size <= IND; size <<= 1) {
        for (int stride = size >> 1; stride > 0; stride >>= 1) {
            for (int p = t; p < IND / 2; p += 256) {
                const int lo = ((p & ~(stride - 1)) << 1) | (p & (stride - 1));
                const int up = lo + stride;
                const bool descBlock = ((lo & size) == 0);
                float va = sv[lo], vb = sv[up];
                int   ia = si[lo], ib = si[up];
                bool aFirst = (va > vb) || (va == vb && ia < ib);
                if (descBlock ? (!aFirst) : aFirst) {
                    sv[lo] = vb; sv[up] = va;
                    si[lo] = ib; si[up] = ia;
                }
            }
            __syncthreads();
        }
    }
    for (int i = t; i < K_; i += 256) {
        tval[b * K_ + i] = sv[i];
        tidx[b * K_ + i] = si[i];
    }
}

// ---------------------------------------------------------------------------
// lb0[b,k,:] = softmax(gelu(tv*pe_w1+pe_b1) @ pe_w2^T + pe_b2) * roles
// One wave per 16 flattened (b,k) rows; WMMA vs pe_w2; softmax in registers.
// ---------------------------------------------------------------------------
__global__ __launch_bounds__(256) void k_lb0(const float* __restrict__ tval,
                                             const int* __restrict__ tidx,
                                             const bf16_t* __restrict__ pew2_bf,
                                             const float* __restrict__ pe_w1,
                                             const float* __restrict__ pe_b1,
                                             const float* __restrict__ pe_b2,
                                             const float* __restrict__ freq,
                                             bf16_t* __restrict__ lb0) {
    __shared__ bf16_t hA[8][16 * 128];
    const int lane = threadIdx.x & 31, w = threadIdx.x >> 5;
    const int ln = lane & 15, hi = lane >> 4;
    const int tile = blockIdx.x * 8 + w;       // 0..28671 (16 rows each)
    const int row0 = tile * 16;                 // tiles never cross b (448%16==0)

    // h tile: lane -> row ln, cols hi*64..hi*64+63
    {
        const float tv = tval[row0 + ln];
        const int jb = hi * 64;
#pragma unroll 8
        for (int j = 0; j < 64; ++j) {
            const int col = jb + j;
            const float z = tv * pe_w1[col] + pe_b1[col];
            hA[w][ln * 128 + col] = (bf16_t)gelu_erf(z);
        }
    }
    wave_lds_fence();

    v8f acc[16];
#pragma unroll
    for (int nt = 0; nt < 16; ++nt)
#pragma unroll
        for (int r = 0; r < 8; ++r) acc[nt][r] = 0.0f;

    for (int ks = 0; ks < 4; ++ks) {
        const int k0 = ks * 32, c0 = k0 + hi * 8, c1 = k0 + 16 + hi * 8;
        v16bf a = frag_ld(&hA[w][ln * 128 + c0], &hA[w][ln * 128 + c1]);
#pragma unroll
        for (int nt = 0; nt < 16; ++nt) {
            const bf16_t* brow = pew2_bf + (nt * 16 + ln) * 128;
            v16bf bm = frag_ld(brow + c0, brow + c1);
            acc[nt] = __builtin_amdgcn_wmma_f32_16x16x32_bf16(false, a, false, bm,
                                                              (short)0, acc[nt], false, false);
        }
    }
#pragma unroll
    for (int nt = 0; nt < 16; ++nt) {
        const float b2 = pe_b2[nt * 16 + ln];
#pragma unroll
        for (int r = 0; r < 8; ++r) acc[nt][r] += b2;
    }

    // Row softmax stats (row = r + hi*8), reduce across 16 lanes of half-wave.
    float rmax[8], rsum[8];
#pragma unroll
    for (int r = 0; r < 8; ++r) {
        float m = -3.0e38f;
#pragma unroll
        for (int nt = 0; nt < 16; ++nt) m = fmaxf(m, acc[nt][r]);
        for (int off = 8; off >= 1; off >>= 1) m = fmaxf(m, __shfl_xor(m, off, 32));
        rmax[r] = m;
        float s = 0.0f;
#pragma unroll
        for (int nt = 0; nt < 16; ++nt) s += __expf(acc[nt][r] - m);
        for (int off = 8; off >= 1; off >>= 1) s += __shfl_xor(s, off, 32);
        rsum[r] = 1.0f / s;
    }
    int idxr[8];
#pragma unroll
    for (int r = 0; r < 8; ++r) idxr[r] = tidx[row0 + r + hi * 8];

#pragma unroll 1
    for (int nt = 0; nt < 16; ++nt) {
        const int col = nt * 16 + ln;
        const float fr = (col < 128) ? freq[col] : freq[col - 128];
#pragma unroll
        for (int r = 0; r < 8; ++r) {
            const float ph = (float)idxr[r] * fr;
            const float role = (col < 128) ? __sinf(ph) : __cosf(ph);
            const float f = __expf(acc[nt][r] - rmax[r]) * rsum[r];
            lb0[(row0 + r + hi * 8) * D_ + col] = (bf16_t)(f * role);
        }
    }
}

// ---------------------------------------------------------------------------
// q = LN(state) @ wq^T + bq      (one block per batch row)
// ---------------------------------------------------------------------------
__global__ __launch_bounds__(256) void k_qproj(const float* __restrict__ state,
                                               const float* __restrict__ attn_in_w,
                                               const float* __restrict__ attn_in_b,
                                               const float* __restrict__ ng,
                                               const float* __restrict__ nb,
                                               float* __restrict__ q) {
    __shared__ float red[256];
    __shared__ float qn[256];
    const int b = blockIdx.x, t = threadIdx.x;
    const float sv = state[b * D_ + t];
    const float mu = block_reduce_sum(sv, red) * (1.0f / 256.0f);
    const float dv = sv - mu;
    const float var = block_reduce_sum(dv * dv, red) * (1.0f / 256.0f);
    qn[t] = dv * rsqrtf(var + 1e-5f) * ng[t] + nb[t];
    __syncthreads();
    const float* wrow = attn_in_w + t * D_;     // wq row t
    float a = attn_in_b[t];
    for (int j = 0; j < D_; ++j) a += qn[j] * wrow[j];
    q[b * D_ + t] = a;
}

// ---------------------------------------------------------------------------
// Main attention kernel, one block per batch row:
//   kn = LN(lb0 * S); khat = kn@wk^T (WMMA); scores; softmax per head;
//   u[h] = attn[h] @ kn; o = u@wv^T + bv; o2 = o@attn_out^T + bo
// ---------------------------------------------------------------------------
__global__ __launch_bounds__(256) void k_attn(const bf16_t* __restrict__ lb0,
                                              const float* __restrict__ Sg,
                                              const float* __restrict__ qg,
                                              const bf16_t* __restrict__ wk_bf,
                                              const float* __restrict__ attn_in_w,
                                              const float* __restrict__ attn_in_b,
                                              const float* __restrict__ attn_out_w,
                                              const float* __restrict__ attn_out_b,
                                              const float* __restrict__ ng,
                                              const float* __restrict__ nb,
                                              float* __restrict__ o2) {
    __shared__ bf16_t knst[8][16 * 256];        // per-wave kn tile staging (64 KB)
    __shared__ float scores[H_][K_];            // scores -> attn weights
    __shared__ float rowMu[K_], rowRs[K_];
    __shared__ float Ssh[256], gsh[256], bsh[256], qsh[256];
    __shared__ float u[H_][256];
    __shared__ float olds[256];

    const int b = blockIdx.x, t = threadIdx.x;
    const int lane = t & 31, w = t >> 5;
    const int ln = lane & 15, hi = lane >> 4;
    const float rscale = 0.17677669529663687f;  // 1/sqrt(32)

    Ssh[t] = Sg[b * D_ + t]; gsh[t] = ng[t]; bsh[t] = nb[t];
    qsh[t] = qg[b * D_ + t];
    for (int i = t; i < H_ * K_; i += 256) (&scores[0][0])[i] = 0.0f;
    __syncthreads();

    // ----- phase 1: per k-tile (16 rows): LN + WMMA k-projection + scores
    for (int kt = w; kt < K_ / 16; kt += 8) {
        const int kr0 = kt * 16;
        const bf16_t* lrow = lb0 + (b * K_ + kr0 + ln) * D_ + hi * 128;
        if (kt + 8 < K_ / 16)
            __builtin_prefetch(lb0 + (b * K_ + kr0 + 128 + ln) * D_ + hi * 128, 0, 1);

        float s1 = 0.0f, s2 = 0.0f;
        for (int jc = 0; jc < 16; ++jc) {
            Chunk8 c; c.q = *reinterpret_cast<const uint4*>(lrow + jc * 8);
#pragma unroll
            for (int e = 0; e < 8; ++e) {
                const int col = hi * 128 + jc * 8 + e;
                const float v = (float)c.e[e] * Ssh[col];
                knst[w][ln * 256 + col] = (bf16_t)v;
                s1 += v; s2 += v * v;
            }
        }
        s1 += __shfl_xor(s1, 16, 32);
        s2 += __shfl_xor(s2, 16, 32);
        const float mu = s1 * (1.0f / 256.0f);
        const float rs = rsqrtf(s2 * (1.0f / 256.0f) - mu * mu + 1e-5f);
        if (hi == 0) { rowMu[kr0 + ln] = mu; rowRs[kr0 + ln] = rs; }
        wave_lds_fence();
#pragma unroll 8
        for (int j = 0; j < 128; ++j) {
            const int col = hi * 128 + j;
            const float v = (float)knst[w][ln * 256 + col];
            knst[w][ln * 256 + col] = (bf16_t)((v - mu) * rs * gsh[col] + bsh[col]);
        }
        wave_lds_fence();

        // Hoist the 8 A-fragments (kn tile) once; reuse across all 16 N-tiles.
        v16bf afr[8];
#pragma unroll
        for (int ks = 0; ks < 8; ++ks) {
            const int k0 = ks * 32, c0 = k0 + hi * 8, c1 = k0 + 16 + hi * 8;
            afr[ks] = frag_ld(&knst[w][ln * 256 + c0], &knst[w][ln * 256 + c1]);
        }

#pragma unroll 1
        for (int nt = 0; nt < 16; ++nt) {
            v8f acc;
#pragma unroll
            for (int r = 0; r < 8; ++r) acc[r] = 0.0f;
            const bf16_t* brow = wk_bf + (nt * 16 + ln) * 256;
#pragma unroll
            for (int ks = 0; ks < 8; ++ks) {
                const int k0 = ks * 32, c0 = k0 + hi * 8, c1 = k0 + 16 + hi * 8;
                v16bf bm = frag_ld(brow + c0, brow + c1);
                acc = __builtin_amdgcn_wmma_f32_16x16x32_bf16(false, afr[ks], false, bm,
                                                              (short)0, acc, false, false);
            }
            const int col = nt * 16 + ln;
            const float qv = qsh[col];
            const float bk = attn_in_b[D_ + col];
            const int hh = nt >> 1;
#pragma unroll
            for (int r = 0; r < 8; ++r) {
                float p = (acc[r] + bk) * qv;
                for (int off = 8; off >= 1; off >>= 1) p += __shfl_xor(p, off, 32);
                if (ln == 0) scores[hh][kr0 + r + hi * 8] += p * rscale;
            }
        }
    }
    __syncthreads();

    // ----- phase 2: softmax over K per head (wave w -> head w)
    {
        float m = -3.0e38f;
        for (int k = lane; k < K_; k += 32) m = fmaxf(m, scores[w][k]);
        for (int off = 16; off >= 1; off >>= 1) m = fmaxf(m, __shfl_xor(m, off, 32));
        float s = 0.0f;
        for (int k = lane; k < K_; k += 32) s += __expf(scores[w][k] - m);
        for (int off = 16; off >= 1; off >>= 1) s += __shfl_xor(s, off, 32);
        const float inv = 1.0f / s;
        for (int k = lane; k < K_; k += 32)
            scores[w][k] = __expf(scores[w][k] - m) * inv;
    }
    __syncthreads();

    // ----- phase 3: u[h][j] = sum_k attn[h][k] * kn[k][j]  (kn recomputed)
    {
        float uh[H_];
#pragma unroll
        for (int h = 0; h < H_; ++h) uh[h] = 0.0f;
        const float Sj = Ssh[t], gj = gsh[t], bj = bsh[t];
        const bf16_t* col = lb0 + b * K_ * D_ + t;
        for (int k = 0; k < K_; ++k) {
            const float v = (float)col[k * D_] * Sj;
            const float kn = (v - rowMu[k]) * rowRs[k] * gj + bj;
#pragma unroll
            for (int h = 0; h < H_; ++h) uh[h] += scores[h][k] * kn;
        }
#pragma unroll
        for (int h = 0; h < H_; ++h) u[h][t] = uh[h];
    }
    __syncthreads();

    // ----- phase 4a: o[d] = u[head(d)] . wv[d] + bv[d]
    {
        const int h = t >> 5;
        const float* wv = attn_in_w + (2 * D_ + t) * D_;
        float a = attn_in_b[2 * D_ + t];
        for (int j = 0; j < D_; ++j) a += u[h][j] * wv[j];
        olds[t] = a;
    }
    __syncthreads();

    // ----- phase 4b: o2 = o @ attn_out^T + bo
    {
        const float* wrow = attn_out_w + t * D_;
        float a = attn_out_b[t];
        for (int j = 0; j < D_; ++j) a += olds[j] * wrow[j];
        o2[b * D_ + t] = a;
    }
}

// ---------------------------------------------------------------------------
// Tail: LN(o2) -> FFN -> gate -> state update -> fb -> S update
// ---------------------------------------------------------------------------
__global__ __launch_bounds__(256) void k_tail(const float* __restrict__ o2,
                                              float* __restrict__ state,
                                              float* __restrict__ S,
                                              const float* __restrict__ ng,
                                              const float* __restrict__ nb,
                                              const float* __restrict__ ffn_w1,
                                              const float* __restrict__ ffn_b1,
                                              const float* __restrict__ ffn_w2,
                                              const float* __restrict__ ffn_b2,
                                              const float* __restrict__ gate_w,
                                              const float* __restrict__ gate_b,
                                              const float* __restrict__ fb_w,
                                              const float* __restrict__ fb_b,
                                              const float* __restrict__ fb_g,
                                              const float* __restrict__ fb_beta) {
    __shared__ float red[256];
    __shared__ float on[256];
    __shared__ float ff1[2 * D_];
    __shared__ float ns[256];
    const int b = blockIdx.x, t = threadIdx.x;
    const float ov = o2[b * D_ + t];
    const float sv = state[b * D_ + t];

    const float mu = block_reduce_sum(ov, red) * (1.0f / 256.0f);
    const float dv = ov - mu;
    const float var = block_reduce_sum(dv * dv, red) * (1.0f / 256.0f);
    on[t] = dv * rsqrtf(var + 1e-5f) * ng[t] + nb[t];
    __syncthreads();

    for (int e = t; e < 2 * D_; e += 256) {
        const float* wr = ffn_w1 + e * D_;
        float a = ffn_b1[e];
        for (int j = 0; j < D_; ++j) a += on[j] * wr[j];
        ff1[e] = gelu_erf(a);
    }
    __syncthreads();

    float ff = ffn_b2[t];
    {
        const float* wr = ffn_w2 + t * (2 * D_);
        for (int j = 0; j < 2 * D_; ++j) ff += ff1[j] * wr[j];
    }
    const float gsum = block_reduce_sum(sv * gate_w[t], red);
    const float gate = 1.0f / (1.0f + __expf(-(gsum + gate_b[0])));
    const float nsv = sv + ff * gate;
    state[b * D_ + t] = nsv;
    ns[t] = nsv;
    __syncthreads();

    float z = fb_b[t];
    {
        const float* wr = fb_w + t * D_;
        for (int j = 0; j < D_; ++j) z += ns[j] * wr[j];
    }
    const float mu2 = block_reduce_sum(z, red) * (1.0f / 256.0f);
    const float dz = z - mu2;
    const float var2 = block_reduce_sum(dz * dz, red) * (1.0f / 256.0f);
    const float fbv = tanhf(dz * rsqrtf(var2 + 1e-5f) * fb_g[t] + fb_beta[t]);
    S[b * D_ + t] *= (1.0f + fbv);
}

// ---------------------------------------------------------------------------
// Head: out = LN(state; head_g, head_b) @ head_w^T + head_bias
// ---------------------------------------------------------------------------
__global__ __launch_bounds__(256) void k_head(const float* __restrict__ state,
                                              const float* __restrict__ hg,
                                              const float* __restrict__ hb,
                                              const float* __restrict__ head_w,
                                              const float* __restrict__ head_bias,
                                              float* __restrict__ out) {
    __shared__ float red[256];
    __shared__ float hn[256];
    const int b = blockIdx.x, t = threadIdx.x;
    const float sv = state[b * D_ + t];
    const float mu = block_reduce_sum(sv, red) * (1.0f / 256.0f);
    const float dv = sv - mu;
    const float var = block_reduce_sum(dv * dv, red) * (1.0f / 256.0f);
    hn[t] = dv * rsqrtf(var + 1e-5f) * hg[t] + hb[t];
    __syncthreads();
    if (t < OUT_) {
        const float* wr = head_w + t * D_;
        float a = head_bias[t];
        for (int j = 0; j < D_; ++j) a += hn[j] * wr[j];
        out[b * OUT_ + t] = a;
    }
}

// ---------------------------------------------------------------------------
extern "C" void kernel_launch(void* const* d_in, const int* in_sizes, int n_in,
                              void* d_out, int out_size, void* d_ws, size_t ws_size,
                              hipStream_t stream) {
    const float* x          = (const float*)d_in[0];
    const float* basis      = (const float*)d_in[1];
    const float* pe_w1      = (const float*)d_in[2];
    const float* pe_b1      = (const float*)d_in[3];
    const float* pe_w2      = (const float*)d_in[4];
    const float* pe_b2      = (const float*)d_in[5];
    const float* freq       = (const float*)d_in[6];
    const float* attn_in_w  = (const float*)d_in[7];
    const float* attn_in_b  = (const float*)d_in[8];
    const float* attn_out_w = (const float*)d_in[9];
    const float* attn_out_b = (const float*)d_in[10];
    const float* norm_g     = (const float*)d_in[11];
    const float* norm_b     = (const float*)d_in[12];
    const float* gate_w     = (const float*)d_in[13];
    const float* gate_b     = (const float*)d_in[14];
    const float* ffn_w1     = (const float*)d_in[15];
    const float* ffn_b1     = (const float*)d_in[16];
    const float* ffn_w2     = (const float*)d_in[17];
    const float* ffn_b2     = (const float*)d_in[18];
    const float* fb_w       = (const float*)d_in[19];
    const float* fb_b       = (const float*)d_in[20];
    const float* fb_g       = (const float*)d_in[21];
    const float* fb_beta    = (const float*)d_in[22];
    const float* head_g     = (const float*)d_in[23];
    const float* head_b_ln  = (const float*)d_in[24];
    const float* head_w     = (const float*)d_in[25];
    const float* head_bias  = (const float*)d_in[26];
    float* out = (float*)d_out;

    char* p = (char*)d_ws;
    auto alloc = [&](size_t bytes) {
        char* r = p;
        p += (bytes + 255) & ~(size_t)255;
        return r;
    };
    float*  xl      = (float*)alloc(sizeof(float) * B_ * IND);
    bf16_t* xlb     = (bf16_t*)alloc(sizeof(bf16_t) * B_ * IND);
    bf16_t* basisT  = (bf16_t*)alloc(sizeof(bf16_t) * IND * D_);
    bf16_t* wk_bf   = (bf16_t*)alloc(sizeof(bf16_t) * D_ * D_);
    bf16_t* pew2_bf = (bf16_t*)alloc(sizeof(bf16_t) * D_ * (D_ / 2));
    float*  tval    = (float*)alloc(sizeof(float) * B_ * K_);
    int*    tidx    = (int*)alloc(sizeof(int) * B_ * K_);
    float*  state   = (float*)alloc(sizeof(float) * B_ * D_);
    float*  S       = (float*)alloc(sizeof(float) * B_ * D_);
    float*  q       = (float*)alloc(sizeof(float) * B_ * D_);
    float*  o2      = (float*)alloc(sizeof(float) * B_ * D_);
    bf16_t* lb0     = (bf16_t*)alloc(sizeof(bf16_t) * (size_t)B_ * K_ * D_);

    k_convert<<<512, 256, 0, stream>>>(basis, attn_in_w, pe_w2, basisT, wk_bf, pew2_bf);
    k_prep<<<1024, 256, 0, stream>>>(x, xl, xlb);
    k_turbo<<<B_ / 16, 256, 0, stream>>>(xlb, basisT, state, S);
    k_topk<<<B_, 256, 0, stream>>>(xl, tval, tidx);
    k_lb0<<<(B_ * K_ / 16) / 8, 256, 0, stream>>>(tval, tidx, pew2_bf, pe_w1, pe_b1,
                                                  pe_b2, freq, lb0);
    for (int it = 0; it < LOOPS; ++it) {
        k_qproj<<<B_, 256, 0, stream>>>(state, attn_in_w, attn_in_b, norm_g, norm_b, q);
        k_attn<<<B_, 256, 0, stream>>>(lb0, S, q, wk_bf, attn_in_w, attn_in_b,
                                       attn_out_w, attn_out_b, norm_g, norm_b, o2);
        k_tail<<<B_, 256, 0, stream>>>(o2, state, S, norm_g, norm_b, ffn_w1, ffn_b1,
                                       ffn_w2, ffn_b2, gate_w, gate_b, fb_w, fb_b,
                                       fb_g, fb_beta);
    }
    k_head<<<B_, 256, 0, stream>>>(state, head_g, head_b_ln, head_w, head_bias, out);
}
